// ProbSparseAttention_34411277975911
// MI455X (gfx1250) — matrix operand order repaired and checked by reading, not compiled, not run on hardware
//
#include <hip/hip_runtime.h>

// ---------------- problem constants (from reference) ----------------
#define BB   4
#define LL   2048
#define DD   512
#define HH   8
#define HD   64
#define UU   8          // ceil(log(L+1)) = 8
#define BH   (BB*HH)    // 32
#define SCALE 0.125f    // 1/sqrt(64)

typedef __attribute__((ext_vector_type(16))) __bf16 v16bf;
typedef __attribute__((ext_vector_type(8)))  __bf16 v8bf;
typedef __attribute__((ext_vector_type(8)))  float  v8f;
typedef __attribute__((ext_vector_type(4)))  float  vf4;

static __device__ __forceinline__ __bf16 f2bf(float f) {
  union { float f; unsigned u; } x; x.f = f;
  unsigned r = x.u + 0x7FFFu + ((x.u >> 16) & 1u);   // round-to-nearest-even
  unsigned short hs = (unsigned short)(r >> 16);
  return __builtin_bit_cast(__bf16, hs);
}

// ---------------------------------------------------------------
// K0: zero the 536MB attn_weights_full region with NT 128-bit stores
// ---------------------------------------------------------------
__global__ void k0_zero_aw(float* __restrict__ aw) {
  size_t i = (size_t)blockIdx.x * blockDim.x + threadIdx.x;
  vf4 z = {0.f, 0.f, 0.f, 0.f};
  __builtin_nontemporal_store(z, ((vf4*)aw) + i);
}

// ---------------------------------------------------------------
// Kc: elementwise fp32 -> bf16 conversion (for Q, K, W)
// ---------------------------------------------------------------
__global__ void kc_cvt(const float* __restrict__ in, __bf16* __restrict__ out) {
  size_t i = (size_t)blockIdx.x * blockDim.x + threadIdx.x;
  out[i] = f2bf(in[i]);
}

// ---------------------------------------------------------------
// K1: M_measure via bf16 WMMA, fragments loaded straight from global
//     (bf16 head-rows are contiguous: A = 2x16B chunks, B = 1x32B chunk).
//     Branchless per-lane online logsumexp, merged across 16 lanes.
// ---------------------------------------------------------------
__global__ void k1_measure(const __bf16* __restrict__ qbf,
                           const __bf16* __restrict__ kbf,
                           float* __restrict__ m_out) {
  const int tiles_m = LL / 16;                 // 128
  int bh = blockIdx.x / tiles_m;
  int mt = blockIdx.x % tiles_m;
  int b = bh / HH, h = bh % HH;
  int lane = threadIdx.x;
  int row  = lane & 15;
  int hi   = (lane >> 4) & 1;
  int m0   = mt * 16;

  // A fragments (row = lane&15): a0 covers k 0..31, a1 covers 32..63
  const __bf16* qrow = qbf + ((size_t)(b * LL + m0 + row)) * DD + h * HD;
  v16bf a0, a1;
  {
    v8bf lo0 = *(const v8bf*)(qrow + hi * 16);           // k = hi*16 .. +8
    v8bf hi0 = *(const v8bf*)(qrow + 8 + hi * 16);       // k = 8+hi*16 .. +8
    v8bf lo1 = *(const v8bf*)(qrow + 32 + hi * 16);
    v8bf hi1 = *(const v8bf*)(qrow + 40 + hi * 16);
#pragma unroll
    for (int e = 0; e < 8; ++e) {
      a0[e] = lo0[e]; a0[e + 8] = hi0[e];
      a1[e] = lo1[e]; a1[e + 8] = hi1[e];
    }
  }

  float lmax[8], lse[8], lsum[8];
#pragma unroll
  for (int r = 0; r < 8; ++r) { lmax[r] = -1e30f; lse[r] = 0.f; lsum[r] = 0.f; }

  const __bf16* kbase = kbf + ((size_t)(b * LL + row)) * DD + h * HD + hi * 16;
  const size_t krowstride = (size_t)16 * DD;             // 16 key rows per tile

  for (int nt = 0; nt < LL / 16; ++nt) {
    const __bf16* kr = kbase + (size_t)nt * krowstride;  // col = lane&15
    v16bf b0 = *(const v16bf*)(kr);                      // k 0..31 (split by hi)
    v16bf b1 = *(const v16bf*)(kr + 32);                 // k 32..63
    v8f c = {};
    c = __builtin_amdgcn_wmma_f32_16x16x32_bf16(false, a0, false, b0,
                                                (short)0, c, false, false);
    c = __builtin_amdgcn_wmma_f32_16x16x32_bf16(false, a1, false, b1,
                                                (short)0, c, false, false);
#pragma unroll
    for (int r = 0; r < 8; ++r) {              // branchless online softmax
      float s  = c[r] * SCALE;
      float nm = fmaxf(lmax[r], s);
      lse[r]   = lse[r] * __expf(lmax[r] - nm) + __expf(s - nm);
      lmax[r]  = nm;
      lsum[r] += s;
    }
  }

  // merge online stats across the 16 lanes of each half-wave group
#pragma unroll
  for (int off = 1; off < 16; off <<= 1) {
#pragma unroll
    for (int r = 0; r < 8; ++r) {
      float om  = __shfl_xor(lmax[r], off, 16);
      float ose = __shfl_xor(lse[r],  off, 16);
      float os  = __shfl_xor(lsum[r], off, 16);
      float nm  = fmaxf(lmax[r], om);
      lse[r]  = lse[r] * __expf(lmax[r] - nm) + ose * __expf(om - nm);
      lmax[r] = nm;
      lsum[r] += os;
    }
  }
  if ((lane & 15) == 0) {
    const float logL = __logf((float)LL);
#pragma unroll
    for (int r = 0; r < 8; ++r) {
      int grow = m0 + r + hi * 8;
      m_out[(size_t)bh * LL + grow] =
          lmax[r] + __logf(lse[r]) - logL - lsum[r] * (1.0f / LL);
    }
  }
}

// ---------------------------------------------------------------
// K2: top-8 per (b,h) over 2048 measure values (iterative argmax)
// ---------------------------------------------------------------
__global__ void k2_topk(const float* __restrict__ m_in, int* __restrict__ topk) {
  int bh = blockIdx.x, tid = threadIdx.x;
  __shared__ float sm[LL];
  __shared__ float rv[256];
  __shared__ int   ri[256];
  for (int m = tid; m < LL; m += 256) sm[m] = m_in[(size_t)bh * LL + m];
  __syncthreads();
  for (int it = 0; it < UU; ++it) {
    float bv = -3.4e38f; int bi = 0;
    for (int m = tid; m < LL; m += 256)
      if (sm[m] > bv) { bv = sm[m]; bi = m; }
    rv[tid] = bv; ri[tid] = bi;
    __syncthreads();
    for (int off = 128; off > 0; off >>= 1) {
      if (tid < off && rv[tid + off] > rv[tid]) {
        rv[tid] = rv[tid + off]; ri[tid] = ri[tid + off];
      }
      __syncthreads();
    }
    if (tid == 0) { topk[bh * UU + it] = ri[0]; sm[ri[0]] = -3.4e38f; }
    __syncthreads();
  }
}

// ---------------------------------------------------------------
// K3: v_avg[bh][d] = mean over L of V
// ---------------------------------------------------------------
__global__ void k3_vavg(const float* __restrict__ v, float* __restrict__ vavg) {
  int bh = blockIdx.x, tid = threadIdx.x;
  int b = bh / HH, h = bh % HH;
  int d = tid & 63, seg = tid >> 6;
  float acc = 0.f;
  for (int l = seg; l < LL; l += 4)
    acc += v[((size_t)(b * LL + l)) * DD + h * HD + d];
  __shared__ float red[256];
  red[tid] = acc;
  __syncthreads();
  if (seg == 0)
    vavg[bh * HD + d] =
        (red[d] + red[64 + d] + red[128 + d] + red[192 + d]) * (1.0f / LL);
}

// ---------------------------------------------------------------
// K4: full attention for the 8 active rows of each (b,h).
//     Writes normalized weights straight into attn_weights_full.
// ---------------------------------------------------------------
__global__ void k4_active(const float* __restrict__ q,
                          const float* __restrict__ kmat,
                          const float* __restrict__ v,
                          const int* __restrict__ topk,
                          float* __restrict__ aw,          // [BH, L, L]
                          float* __restrict__ act) {       // [BH, U, HD]
  int bh = blockIdx.x >> 3, j = blockIdx.x & 7;
  int b = bh / HH, h = bh % HH;
  int tid = threadIdx.x;
  int idx = topk[bh * UU + j];

  __shared__ float qsh[HD];
  __shared__ float sc[LL];
  __shared__ float red[256];

  if (tid < HD) qsh[tid] = q[((size_t)(b * LL + idx)) * DD + h * HD + tid];
  __syncthreads();

  for (int m = tid; m < LL; m += 256) {
    const float* kr = &kmat[((size_t)(b * LL + m)) * DD + h * HD];
    float s = 0.f;
#pragma unroll 8
    for (int d = 0; d < HD; ++d) s += qsh[d] * kr[d];
    sc[m] = s * SCALE;
  }
  __syncthreads();

  float lm = -3.4e38f;
  for (int m = tid; m < LL; m += 256) lm = fmaxf(lm, sc[m]);
  red[tid] = lm; __syncthreads();
  for (int off = 128; off > 0; off >>= 1) {
    if (tid < off) red[tid] = fmaxf(red[tid], red[tid + off]);
    __syncthreads();
  }
  float mx = red[0];
  __syncthreads();

  float ls = 0.f;
  for (int m = tid; m < LL; m += 256) {
    float w = __expf(sc[m] - mx);
    sc[m] = w; ls += w;
  }
  red[tid] = ls; __syncthreads();
  for (int off = 128; off > 0; off >>= 1) {
    if (tid < off) red[tid] += red[tid + off];
    __syncthreads();
  }
  float inv = 1.0f / red[0];
  __syncthreads();

  float* awr = aw + (size_t)bh * LL * LL + (size_t)idx * LL;
  for (int m = tid; m < LL; m += 256) awr[m] = sc[m] * inv;

  int d = tid & 63, seg = tid >> 6;
  float acc = 0.f;
  for (int m = seg * 512; m < seg * 512 + 512; ++m)
    acc += sc[m] * v[((size_t)(b * LL + m)) * DD + h * HD + d];
  red[tid] = acc; __syncthreads();
  if (seg == 0)
    act[(bh * UU + j) * HD + d] =
        (red[d] + red[64 + d] + red[128 + d] + red[192 + d]) * inv;
}

// ---------------------------------------------------------------
// K5a: xbf[b,l,h*64+d] = bf16(v_avg)  /  K5b: scatter active rows (bf16)
// ---------------------------------------------------------------
__global__ void k5a_fillx(const float* __restrict__ vavg, __bf16* __restrict__ x) {
  size_t i = (size_t)blockIdx.x * 256 + threadIdx.x;    // over B*L*D
  int dd = (int)(i % DD);
  int b  = (int)(i / ((size_t)LL * DD));
  int h  = dd >> 6, d = dd & 63;
  x[i] = f2bf(vavg[(b * HH + h) * HD + d]);
}

__global__ void k5b_scatterx(const int* __restrict__ topk,
                             const float* __restrict__ act,
                             __bf16* __restrict__ x) {
  int bh = blockIdx.x >> 3, j = blockIdx.x & 7;
  int b = bh / HH, h = bh % HH;
  int idx = topk[bh * UU + j];
  x[((size_t)(b * LL + idx)) * DD + h * HD + threadIdx.x] =
      f2bf(act[(bh * UU + j) * HD + threadIdx.x]);
}

// ---------------------------------------------------------------
// K6: out = x @ W^T + bias   (bf16 WMMA, fragments straight from global)
// ---------------------------------------------------------------
__global__ void k6_gemm(const __bf16* __restrict__ xbf,
                        const __bf16* __restrict__ wbf,
                        const float* __restrict__ bias,
                        float* __restrict__ out) {
  int i0 = (blockIdx.x / (DD / 16)) * 16;      // output row tile
  int j0 = (blockIdx.x % (DD / 16)) * 16;      // output col tile
  int lane = threadIdx.x;
  int row = lane & 15, hi = (lane >> 4) & 1;

  const __bf16* xrow = xbf + (size_t)(i0 + row) * DD;    // A row
  const __bf16* wrow = wbf + (size_t)(j0 + row) * DD;    // B col (= W row)

  v8f c = {};
  for (int kt = 0; kt < DD / 32; ++kt) {
    int kb = kt * 32;
    v8bf alo = *(const v8bf*)(xrow + kb + hi * 16);
    v8bf ahi = *(const v8bf*)(xrow + kb + 8 + hi * 16);
    v16bf a;
#pragma unroll
    for (int e = 0; e < 8; ++e) { a[e] = alo[e]; a[e + 8] = ahi[e]; }
    v16bf bf = *(const v16bf*)(wrow + kb + hi * 16);
    c = __builtin_amdgcn_wmma_f32_16x16x32_bf16(false, a, false, bf,
                                                (short)0, c, false, false);
  }
  int col = lane & 15;
  float bv = bias[j0 + col];
#pragma unroll
  for (int r = 0; r < 8; ++r)
    out[((size_t)(i0 + r + hi * 8)) * DD + j0 + col] = c[r] + bv;
}

// ---------------------------------------------------------------
extern "C" void kernel_launch(void* const* d_in, const int* in_sizes, int n_in,
                              void* d_out, int out_size, void* d_ws, size_t ws_size,
                              hipStream_t stream) {
  const float* q    = (const float*)d_in[0];
  const float* kmat = (const float*)d_in[1];
  const float* v    = (const float*)d_in[2];
  const float* wout = (const float*)d_in[3];
  const float* bout = (const float*)d_in[4];

  float* out = (float*)d_out;                             // [B,L,D]
  float* aw  = out + (size_t)BB * LL * DD;                // [B,H,L,L]

  // workspace layout (bytes)
  const size_t NQ = (size_t)BB * LL * DD;                 // 4,194,304
  char* base = (char*)d_ws;
  __bf16* qbf = (__bf16*)base;                            // 8 MB
  __bf16* kbf = (__bf16*)(base + 2 * NQ);                 // 8 MB
  __bf16* xbf = (__bf16*)(base + 4 * NQ);                 // 8 MB
  __bf16* wbf = (__bf16*)(base + 6 * NQ);                 // 512 KB
  float*  mms = (float*)(base + 6 * NQ + 2 * (size_t)DD * DD);   // 65536 f
  int*   topk = (int*)(mms + (size_t)BH * LL);
  float* vavg = (float*)(topk + BH * UU);
  float* act  = vavg + BH * HD;

  // 1) zero attn_weights_full (536 MB, NT 128-bit stores)
  size_t n4 = ((size_t)BH * LL * LL) / 4;
  k0_zero_aw<<<(unsigned)(n4 / 256), 256, 0, stream>>>(aw);

  // 2) pre-convert Q, K, W to bf16
  kc_cvt<<<(unsigned)(NQ / 256), 256, 0, stream>>>(q, qbf);
  kc_cvt<<<(unsigned)(NQ / 256), 256, 0, stream>>>(kmat, kbf);
  kc_cvt<<<(unsigned)((size_t)DD * DD / 256), 256, 0, stream>>>(wout, wbf);

  // 3) sparsity measure (WMMA)
  k1_measure<<<BH * (LL / 16), 32, 0, stream>>>(qbf, kbf, mms);

  // 4) top-8 rows per (b,h)
  k2_topk<<<BH, 256, 0, stream>>>(mms, topk);

  // 5) v mean per (b,h)
  k3_vavg<<<BH, 256, 0, stream>>>(v, vavg);

  // 6) exact attention on active rows (+ scatter weights into aw)
  k4_active<<<BH * UU, 256, 0, stream>>>(q, kmat, v, topk, aw, act);

  // 7) materialize x (bf16) = scattered attn output in [B,L,D]
  k5a_fillx<<<(BB * LL * DD) / 256, 256, 0, stream>>>(vavg, xbf);
  k5b_scatterx<<<BH * UU, HD, 0, stream>>>(topk, act, xbf);

  // 8) output projection (WMMA)
  k6_gemm<<<(BB * LL / 16) * (DD / 16), 32, 0, stream>>>(xbf, wbf, bout, out);
}
